// RVQAudioQuantizer_11922829214090
// MI455X (gfx1250) — compile-verified, matrix operand order checked by reading, not run on hardware
//
#include <hip/hip_runtime.h>
#include <hip/hip_bf16.h>
#include <stdint.h>

#define B_    8
#define T_    4096
#define D_    1280
#define Q_    4
#define K_    256
#define ROWS  32       // M rows per block (two 16-row WMMA tiles)
#define BLOCK 256      // 8 waves (wave32)
#define WAVES 8

typedef __attribute__((ext_vector_type(16))) __bf16 v16bf;
typedef __attribute__((ext_vector_type(8)))  float  v8f;

union FragU { v16bf v; uint4 q[2]; };

__device__ __forceinline__ unsigned short f32_to_bf16_rne(float f) {
  unsigned u = __float_as_uint(f);
  unsigned r = u + 0x7FFFu + ((u >> 16) & 1u);
  return (unsigned short)(r >> 16);
}

// monotone float->uint mapping; key = (dist, idx): min key == min dist, ties -> lowest idx
__device__ __forceinline__ unsigned long long packKey(float d, unsigned idx) {
  unsigned u = __float_as_uint(d);
  u = (u & 0x80000000u) ? ~u : (u | 0x80000000u);
  return ((unsigned long long)u << 32) | (unsigned long long)idx;
}

// ---------------- prep kernels ----------------

__global__ void rvq_prep_zero(float* commitAcc, float* denomAcc) {
  if (threadIdx.x == 0) { *commitAcc = 0.0f; *denomAcc = 0.0f; }
}

// one block per code: convert f32 codebook row -> bf16, compute ||c||^2
__global__ __launch_bounds__(256)
void rvq_prep_cb(const float* __restrict__ cb, unsigned short* __restrict__ cbw,
                 float* __restrict__ cnorm) {
  __shared__ float red[WAVES];
  const int code = blockIdx.x;                 // [0, Q*K)
  const float* src = cb + (size_t)code * D_;
  unsigned short* dst = cbw + (size_t)code * D_;
  float s = 0.0f;
  for (int d = threadIdx.x; d < D_; d += 256) {
    float v = src[d];
    dst[d] = f32_to_bf16_rne(v);
    s += v * v;
  }
  for (int m = 1; m < 32; m <<= 1) s += __shfl_xor(s, m, 32);
  int wave = threadIdx.x >> 5, lane = threadIdx.x & 31;
  if (lane == 0) red[wave] = s;
  __syncthreads();
  if (threadIdx.x == 0) {
    float t = 0.0f;
    for (int w = 0; w < WAVES; ++w) t += red[w];
    cnorm[code] = t;
  }
}

__global__ __launch_bounds__(256)
void rvq_prep_mask(const int* __restrict__ mask, float* __restrict__ denomAcc) {
  __shared__ float red[WAVES];
  int i = blockIdx.x * 256 + threadIdx.x;
  float v = (i < B_ * T_ && mask[i] != 0) ? 1.0f : 0.0f;
  for (int m = 1; m < 32; m <<= 1) v += __shfl_xor(v, m, 32);
  int wave = threadIdx.x >> 5, lane = threadIdx.x & 31;
  if (lane == 0) red[wave] = v;
  __syncthreads();
  if (threadIdx.x == 0) {
    float t = 0.0f;
    for (int w = 0; w < WAVES; ++w) t += red[w];
    atomicAdd(denomAcc, t);
  }
}

// ---------------- per-stage kernel ----------------
// Block: 32 rows x all 256 codes. 8 waves; wave w -> codes [32w, 32w+32) as two
// 16-wide N tiles, against two 16-row M tiles (2M x 2N register blocking:
// every fetched A/B fragment feeds two WMMAs).
__global__ __launch_bounds__(BLOCK)
void rvq_stage(const float* __restrict__ residIn,
               float* __restrict__ residOut,
               const unsigned short* __restrict__ cbw,   // [K_,D_] bf16, this stage
               const float* __restrict__ cbf,            // [K_,D_] f32,  this stage
               const float* __restrict__ cnorm,          // [K_]
               const int* __restrict__ mask,             // [B*T]
               float* __restrict__ idxOut,               // [B*T*Q_] (as float)
               float* __restrict__ commitAcc,
               int stage) {
  __shared__ unsigned short ldsA[ROWS * D_];             // 80 KB bf16 residual tile
  __shared__ unsigned long long ldsKey[WAVES * ROWS];    // 2 KB
  __shared__ int   ldsIdx[ROWS];
  __shared__ float ldsMf[ROWS];
  __shared__ float ldsRed[WAVES];

  const int tid = threadIdx.x;
  const int rowBase = blockIdx.x * ROWS;
  const size_t gbase = (size_t)rowBase * D_;

  // stage residual rows -> LDS as bf16
  for (int i = tid; i < ROWS * D_; i += BLOCK)
    ldsA[i] = f32_to_bf16_rne(residIn[gbase + i]);
  if (tid < ROWS) ldsMf[tid] = (mask[rowBase + tid] != 0) ? 1.0f : 0.0f;
  __syncthreads();

  const int wave = tid >> 5;
  const int lane = tid & 31;
  const int half = lane >> 4;       // K-group selector (A/B frags)
  const int l15  = lane & 15;       // M within tile (A) / N (B,C frags)
  const int n0   = wave * 32;

  v8f acc00 = {};   // M-tile 0, N-tile 0
  v8f acc01 = {};   // M-tile 0, N-tile 1
  v8f acc10 = {};   // M-tile 1, N-tile 0
  v8f acc11 = {};   // M-tile 1, N-tile 1

  const uint4* ldsA4 = (const uint4*)ldsA;
  const uint4* cb4   = (const uint4*)cbw;
  // per-lane fragment pointers; each frag = two 16B chunks at +0 and +16 bf16
  const uint4* aP0 = ldsA4 + (((l15)      * D_ + half * 8) >> 3);
  const uint4* aP1 = ldsA4 + (((l15 + 16) * D_ + half * 8) >> 3);
  const uint4* bP0 = cb4   + (((n0 + l15)      * D_ + half * 8) >> 3);
  const uint4* bP1 = cb4   + (((n0 + 16 + l15) * D_ + half * 8) >> 3);

  for (int kk = 0; kk < D_; kk += 32) {
    FragU a0, a1, b0, b1;
    a0.q[0] = aP0[0]; a0.q[1] = aP0[2];
    a1.q[0] = aP1[0]; a1.q[1] = aP1[2];
    b0.q[0] = bP0[0]; b0.q[1] = bP0[2];
    b1.q[0] = bP1[0]; b1.q[1] = bP1[2];
    acc00 = __builtin_amdgcn_wmma_f32_16x16x32_bf16(false, a0.v, false, b0.v,
                                                    (short)0, acc00, false, false);
    acc01 = __builtin_amdgcn_wmma_f32_16x16x32_bf16(false, a0.v, false, b1.v,
                                                    (short)0, acc01, false, false);
    acc10 = __builtin_amdgcn_wmma_f32_16x16x32_bf16(false, a1.v, false, b0.v,
                                                    (short)0, acc10, false, false);
    acc11 = __builtin_amdgcn_wmma_f32_16x16x32_bf16(false, a1.v, false, b1.v,
                                                    (short)0, acc11, false, false);
    aP0 += 4; aP1 += 4; bP0 += 4; bP1 += 4;   // advance 32 bf16 in K
  }

  // d2 = ||c||^2 - 2*dot ; argmin over this wave's 32 codes, per row
  const float cn0 = cnorm[n0 + l15];
  const float cn1 = cnorm[n0 + 16 + l15];
  unsigned long long keys0[8];   // rows half*8+v       (M-tile 0)
  unsigned long long keys1[8];   // rows 16+half*8+v    (M-tile 1)
  for (int v = 0; v < 8; ++v) {
    unsigned long long ka = packKey(cn0 - 2.0f * acc00[v], (unsigned)(n0 + l15));
    unsigned long long kb = packKey(cn1 - 2.0f * acc01[v], (unsigned)(n0 + 16 + l15));
    unsigned long long k = ka < kb ? ka : kb;
    unsigned long long kc = packKey(cn0 - 2.0f * acc10[v], (unsigned)(n0 + l15));
    unsigned long long kd = packKey(cn1 - 2.0f * acc11[v], (unsigned)(n0 + 16 + l15));
    unsigned long long k2 = kc < kd ? kc : kd;
    for (int m = 1; m <= 8; m <<= 1) {        // reduce across 16 columns (stays in half)
      unsigned long long o  = __shfl_xor(k,  m, 32);
      unsigned long long o2 = __shfl_xor(k2, m, 32);
      k  = (o  < k ) ? o  : k;
      k2 = (o2 < k2) ? o2 : k2;
    }
    keys0[v] = k;
    keys1[v] = k2;
  }
  if (l15 == 0) {
    for (int v = 0; v < 8; ++v) {
      ldsKey[wave * ROWS + half * 8 + v]      = keys0[v];
      ldsKey[wave * ROWS + 16 + half * 8 + v] = keys1[v];
    }
  }
  __syncthreads();

  // combine across 8 waves -> final argmin per row
  if (tid < ROWS) {
    unsigned long long best = ldsKey[tid];
    for (int w = 1; w < WAVES; ++w) {
      unsigned long long k = ldsKey[w * ROWS + tid];
      best = (k < best) ? k : best;
    }
    int idx = (int)(best & 0xFFFFFFFFull);
    ldsIdx[tid] = idx;
    idxOut[(size_t)(rowBase + tid) * Q_ + stage] = (float)idx;
  }
  __syncthreads();

  // residual update + masked commitment-loss partial
  float cacc = 0.0f;
  for (int i = tid; i < ROWS * D_; i += BLOCK) {
    const int row = i / D_;
    const int d   = i - row * D_;
    const float r  = residIn[gbase + i];
    const float cv = cbf[(size_t)ldsIdx[row] * D_ + d];
    const float diff = r - cv;
    residOut[gbase + i] = diff;
    cacc += ldsMf[row] * diff * diff;
  }
  for (int m = 1; m < 32; m <<= 1) cacc += __shfl_xor(cacc, m, 32);
  if (lane == 0) ldsRed[wave] = cacc;
  __syncthreads();
  if (tid == 0) {
    float s = 0.0f;
    for (int w = 0; w < WAVES; ++w) s += ldsRed[w];
    atomicAdd(commitAcc, s * (1.0f / (float)D_));
  }
}

// ---------------- final kernel ----------------
// quantized_out = z - residual_final (telescoping identity); commit = acc / max(denom,1)
__global__ __launch_bounds__(256)
void rvq_final(const float* __restrict__ z, const float* __restrict__ resid,
               float* __restrict__ quantOut, const float* __restrict__ commitAcc,
               const float* __restrict__ denomAcc, float* __restrict__ commitOut) {
  const size_t n = (size_t)B_ * T_ * D_;
  for (size_t i = (size_t)blockIdx.x * 256 + threadIdx.x; i < n;
       i += (size_t)gridDim.x * 256)
    quantOut[i] = z[i] - resid[i];
  if (blockIdx.x == 0 && threadIdx.x == 0)
    commitOut[0] = commitAcc[0] / fmaxf(denomAcc[0], 1.0f);
}

// ---------------- launch ----------------
extern "C" void kernel_launch(void* const* d_in, const int* in_sizes, int n_in,
                              void* d_out, int out_size, void* d_ws, size_t ws_size,
                              hipStream_t stream) {
  (void)in_sizes; (void)n_in; (void)out_size; (void)ws_size;
  const float* z    = (const float*)d_in[0];          // [B,T,D]
  const int*   mask = (const int*)d_in[1];            // [B,T]
  const float* cb   = (const float*)d_in[2];          // [Q,K,D]

  float* out      = (float*)d_out;
  float* quantOut = out;                              // [B,T,D]
  float* idxOut   = out + (size_t)B_ * T_ * D_;       // [B,T,Q]
  float* commitOut = idxOut + (size_t)B_ * T_ * Q_;   // scalar

  char* ws = (char*)d_ws;
  float* resid = (float*)ws;                 ws += (size_t)B_ * T_ * D_ * sizeof(float);
  unsigned short* cbw = (unsigned short*)ws; ws += (size_t)Q_ * K_ * D_ * sizeof(unsigned short);
  float* cnorm = (float*)ws;                 ws += (size_t)Q_ * K_ * sizeof(float);
  float* commitAcc = (float*)ws;             ws += sizeof(float);
  float* denomAcc = (float*)ws;

  rvq_prep_zero<<<1, 32, 0, stream>>>(commitAcc, denomAcc);
  rvq_prep_cb<<<Q_ * K_, 256, 0, stream>>>(cb, cbw, cnorm);
  rvq_prep_mask<<<(B_ * T_ + 255) / 256, 256, 0, stream>>>(mask, denomAcc);

  for (int s = 0; s < Q_; ++s) {
    const float* rin = (s == 0) ? z : resid;
    rvq_stage<<<(B_ * T_) / ROWS, BLOCK, 0, stream>>>(
        rin, resid,
        cbw + (size_t)s * K_ * D_,
        cb  + (size_t)s * K_ * D_,
        cnorm + (size_t)s * K_,
        mask, idxOut, commitAcc, s);
  }

  rvq_final<<<2048, 256, 0, stream>>>(z, resid, quantOut, commitAcc, denomAcc, commitOut);
}
